// LinTranUnetDecoder_63728724738180
// MI455X (gfx1250) — matrix operand order repaired and checked by reading, not compiled, not run on hardware
//
#include <hip/hip_runtime.h>
#include <hip/hip_bf16.h>
#include <math.h>
#include <cstdint>

// ---------------------------------------------------------------------------
// LinTranUnetDecoder for MI455X (gfx1250, wave32, WMMA).
// GEMMs: v_wmma_f32_16x16x32_bf16 with double-buffered LDS tiles:
//   A tile:  per-lane GLOBAL_LOAD_ASYNC_TO_LDS_B128   (ASYNCcnt)
//   B tile:  one TDM tensor_load_to_lds 2D descriptor (TENSORcnt), wave 0
// ---------------------------------------------------------------------------

typedef __bf16 bf16;
typedef __attribute__((ext_vector_type(16))) bf16  v16bf;
typedef __attribute__((ext_vector_type(8)))  bf16  v8bf;
typedef __attribute__((ext_vector_type(8)))  float v8f;
typedef __attribute__((ext_vector_type(4)))  unsigned int u32x4;
typedef __attribute__((ext_vector_type(4)))  int  i32x4;
typedef __attribute__((ext_vector_type(8)))  int  i32x8;

#define NBATCH 2
#define NTOK   13824          // 24*24*24
#define TBTOT  27648          // NBATCH * NTOK
#define DMODEL 256
#define DFFN   512
#define NHEAD  8
#define DKH    32

#if defined(__has_builtin)
#  if __has_builtin(__builtin_amdgcn_tensor_load_to_lds)
#    define HAVE_TDM 1
#  else
#    define HAVE_TDM 0
#  endif
#  if __has_builtin(__builtin_amdgcn_s_wait_tensorcnt)
#    define HAVE_WAIT_TENSOR 1
#  else
#    define HAVE_WAIT_TENSOR 0
#  endif
#else
#  define HAVE_TDM 0
#  define HAVE_WAIT_TENSOR 0
#endif

static __device__ __forceinline__ void wait_tensorcnt0()
{
#if HAVE_WAIT_TENSOR
    __builtin_amdgcn_s_wait_tensorcnt(0);
#else
    asm volatile("s_wait_tensorcnt 0" ::: "memory");
#endif
}

#if HAVE_TDM
// One 2D TDM tile load: Kelems-wide rows (bf16), tile 32 x Nrows(=64),
// LDS rows padded 64B data + 16B pad  (pad_interval=3 -> 64B, pad_amount=3 -> 16B).
static __device__ __forceinline__ void tdm_load_tile_2d(
    unsigned lds_addr, unsigned long long gaddr, int Kelems, int Nrows)
{
    u32x4 g0;
    g0.x = 1u;                                        // count=1 (valid descriptor)
    g0.y = lds_addr;                                  // lds_addr [63:32]
    g0.z = (unsigned)gaddr;                           // global_addr [95:64]
    g0.w = (unsigned)(gaddr >> 32) | (2u << 30);      // global_addr hi | type=2
    i32x8 g1;
    g1[0] = (1 << 16) | (1 << 20) | (3 << 22) | (3 << 25);
    //       data_size=2B | pad_en | pad_interval=64B | pad_amount=16B
    g1[1] = (int)((Kelems & 0xFFFF) << 16);           // tensor_dim0[15:0] @ [63:48]
    g1[2] = (int)(((unsigned)Kelems >> 16) | ((unsigned)(Nrows & 0xFFFF) << 16));
    g1[3] = (int)(((unsigned)Nrows >> 16) | (32u << 16));   // tile_dim0 = 32 elems
    g1[4] = 64;                                       // tile_dim1 = 64 rows, tile_dim2 = 0
    g1[5] = Kelems;                                   // tensor_dim0_stride low 32
    g1[6] = 0;
    g1[7] = 0;
    i32x4 gz = {0, 0, 0, 0};
#if defined(__clang_major__) && (__clang_major__ >= 23)
    i32x8 gz8 = {0, 0, 0, 0, 0, 0, 0, 0};
    __builtin_amdgcn_tensor_load_to_lds(g0, g1, gz, gz, gz8, 0);
#else
    __builtin_amdgcn_tensor_load_to_lds(g0, g1, gz, gz, 0);
#endif
}
#endif

// ---------------------------------------------------------------------------
// Generic bf16 WMMA GEMM:  C[M,N] = act(A[M,K] * Wt[N,K]^T + bias)
//   A  : [M, K] row-major bf16 (activations)  -> async per-lane copies
//   Wt : [N, K] row-major bf16 (weights)      -> TDM descriptor tile loads
// Block = 128 threads (4 waves), 64x64 output tile, K step 32, 2-deep buffer.
// ACT: 0 = none, 1 = relu, 2 = gelu (exact, erf-based)
// ---------------------------------------------------------------------------
template <int ACT>
__global__ __launch_bounds__(128)
void gemm_bf16_wmma(const bf16* __restrict__ A, const bf16* __restrict__ Wt,
                    const float* __restrict__ bias,
                    float* __restrict__ Cf, bf16* __restrict__ Cb,
                    int M, int N, int K)
{
    __shared__ bf16 As[2][64][40];   // [buf][m][k], 80 B rows (64 B data + 16 B pad)
    __shared__ bf16 Bt[2][64][40];   // [buf][n][k], same padding (matches TDM pad)

    const int tid  = threadIdx.x;
    const int wave = tid >> 5;
    const int lane = tid & 31;
    const int m0   = blockIdx.x * 64;
    const int n0   = blockIdx.y * 64;

    v8f acc[4];
#pragma unroll
    for (int j = 0; j < 4; ++j)
        acc[j] = (v8f){0.f, 0.f, 0.f, 0.f, 0.f, 0.f, 0.f, 0.f};

    // A-tile async mapping: each thread moves 2 x 16 B
    const int seg = tid & 3;          // 16-byte chunk of a 64-byte row
    const int rw0 = tid >> 2;         // rows rw0 and rw0+32

    auto issue_tile = [&](int buf, int k0) {
        // ---- A tile: per-lane async global->LDS (ASYNCcnt) ----
#pragma unroll
        for (int rr = 0; rr < 2; ++rr) {
            const int r = rw0 + rr * 32;
            unsigned lA = (unsigned)(uintptr_t)&As[buf][r][seg * 8];
            unsigned long long gA =
                (unsigned long long)(A + (size_t)(m0 + r) * K + k0 + seg * 8);
            asm volatile("global_load_async_to_lds_b128 %0, %1, off"
                         :: "v"(lA), "v"(gA) : "memory");
        }
        // ---- B tile: one TDM 2D tile descriptor (TENSORcnt), wave 0 only ----
#if HAVE_TDM
        if (wave == 0) {
            unsigned lB = (unsigned)(uintptr_t)&Bt[buf][0][0];
            unsigned long long gB =
                (unsigned long long)(Wt + (size_t)n0 * K + k0);
            tdm_load_tile_2d(lB, gB, K, N < 64 ? N : 64);
        }
#else
#pragma unroll
        for (int rr = 0; rr < 2; ++rr) {
            const int r = rw0 + rr * 32;
            unsigned lB = (unsigned)(uintptr_t)&Bt[buf][r][seg * 8];
            unsigned long long gB =
                (unsigned long long)(Wt + (size_t)(n0 + r) * K + k0 + seg * 8);
            asm volatile("global_load_async_to_lds_b128 %0, %1, off"
                         :: "v"(lB), "v"(gB) : "memory");
        }
#endif
        // prefetch the tile after this one toward L2 (speculative)
        if (k0 + 32 < K) {
            __builtin_prefetch(A + (size_t)(m0 + rw0) * K + k0 + 32, 0, 0);
            __builtin_prefetch(Wt + (size_t)(n0 + rw0) * K + k0 + 32, 0, 0);
        }
    };

    const int nk = K >> 5;
    issue_tile(0, 0);

    for (int kt = 0; kt < nk; ++kt) {
        // wait for tile kt (issued one iteration ago; overlapped with math)
        asm volatile("s_wait_asynccnt 0" ::: "memory");
        wait_tensorcnt0();                    // no-op for non-issuing waves
        __syncthreads();                      // publish tile to all waves
        if (kt + 1 < nk) issue_tile((kt + 1) & 1, (kt + 1) * 32);

        const int buf = kt & 1;
        // ---- A fragment (16x32 bf16, ISA 7.12.2 wave32 layout) ----
        const int mrow = wave * 16 + (lane & 15);
        const int kh   = (lane >> 4) * 8;   // lanes 0-15: K 0..7/16..23; 16-31: 8..15/24..31
        v8bf alo = *reinterpret_cast<const v8bf*>(&As[buf][mrow][kh]);
        v8bf ahi = *reinterpret_cast<const v8bf*>(&As[buf][mrow][16 + kh]);
        v16bf afrag = __builtin_shufflevector(alo, ahi,
            0,1,2,3,4,5,6,7,8,9,10,11,12,13,14,15);

        const int kh2 = (lane >> 4) * 8;    // B: lanes 0-15 hold K 0..7/16..23, 16-31 hold 8..15/24..31
#pragma unroll
        for (int j = 0; j < 4; ++j) {
            const bf16* brp = &Bt[buf][j * 16 + (lane & 15)][kh2];
            v8bf blo = *reinterpret_cast<const v8bf*>(brp);
            v8bf bhi = *reinterpret_cast<const v8bf*>(brp + 16);
            v16bf bfrag = __builtin_shufflevector(blo, bhi,
                0,1,2,3,4,5,6,7,8,9,10,11,12,13,14,15);
            acc[j] = __builtin_amdgcn_wmma_f32_16x16x32_bf16(
                false, afrag, false, bfrag, (short)0, acc[j], false, false);
        }
        __syncthreads();   // everyone done reading buf before it is re-filled
    }

    // ---- epilogue: bias + activation, dual f32/bf16 stores ----
    const int mbase = m0 + wave * 16 + (lane >> 4) * 8;  // VGPR r: rows r / r+8
    const int nlane = lane & 15;
#pragma unroll
    for (int j = 0; j < 4; ++j) {
        const int gn = n0 + j * 16 + nlane;
        const float bv = bias ? bias[gn] : 0.f;
#pragma unroll
        for (int r = 0; r < 8; ++r) {
            float y = acc[j][r] + bv;
            if (ACT == 1) y = fmaxf(y, 0.f);
            else if (ACT == 2) y = 0.5f * y * (1.f + erff(y * 0.70710678118f));
            const size_t idx = (size_t)(mbase + r) * N + gn;
            if (Cf) Cf[idx] = y;
            if (Cb) Cb[idx] = (bf16)y;
        }
    }
}

// ---------------------------------------------------------------------------
// Small helper kernels
// ---------------------------------------------------------------------------
__global__ void cast_f32_bf16(const float* __restrict__ in, bf16* __restrict__ out, int n)
{
    int i = blockIdx.x * 256 + threadIdx.x;
    if (i < n) out[i] = (bf16)in[i];
}

// in: [K, N] f32 row-major -> out: [N, K] bf16 (weight pre-transpose)
__global__ void cast_transpose(const float* __restrict__ in, bf16* __restrict__ out,
                               int Kd, int Nd)
{
    int i = blockIdx.x * 256 + threadIdx.x;
    if (i < Kd * Nd) {
        int k = i / Nd, n = i - k * Nd;
        out[(size_t)n * Kd + k] = (bf16)in[i];
    }
}

// x [b, 128, n] f32 -> xt [b*n, 128] bf16 (token-major A matrix)
__global__ void transpose_cast_x(const float* __restrict__ x, bf16* __restrict__ xt)
{
    int i  = blockIdx.x * 256 + threadIdx.x;  // TBTOT*128
    int c  = i & 127;
    int nn = i >> 7;                          // b*NTOK + n
    int b  = nn / NTOK, n = nn - b * NTOK;
    xt[(size_t)nn * 128 + c] = (bf16)x[((size_t)b * 128 + c) * NTOK + n];
}

// softmax over dk (32) per (token, head); wave32 shuffle reduction; * 1/sqrt(dk)
__global__ void softmax_q_kernel(float* __restrict__ q)
{
    int g    = blockIdx.x * 8 + (threadIdx.x >> 5);  // token-head group
    int lane = threadIdx.x & 31;
    int bt   = g >> 3, h = g & 7;
    float* p = q + (size_t)bt * DMODEL + h * DKH;
    float x = p[lane];
    float m = x;
    for (int off = 16; off; off >>= 1) m = fmaxf(m, __shfl_xor(m, off, 32));
    float e = expf(x - m);
    float s = e;
    for (int off = 16; off; off >>= 1) s += __shfl_xor(s, off, 32);
    p[lane] = e / s * 0.17677669529f;   // 1/sqrt(32)
}

// softmax over the token axis (13824) per (b, h, dk) column, in place
__global__ __launch_bounds__(256)
void softmax_k_kernel(float* __restrict__ k)
{
    __shared__ float red[256];
    int col = blockIdx.x;                 // 0..511  (b*256 + channel)
    int b = col >> 8, c = col & 255;
    int tid = threadIdx.x;
    float* base = k + (size_t)b * NTOK * DMODEL + c;

    float mx = -1e30f;
    for (int n = tid; n < NTOK; n += 256) mx = fmaxf(mx, base[(size_t)n * DMODEL]);
    red[tid] = mx; __syncthreads();
    for (int s = 128; s; s >>= 1) { if (tid < s) red[tid] = fmaxf(red[tid], red[tid + s]); __syncthreads(); }
    mx = red[0]; __syncthreads();

    float sum = 0.f;
    for (int n = tid; n < NTOK; n += 256) sum += expf(base[(size_t)n * DMODEL] - mx);
    red[tid] = sum; __syncthreads();
    for (int s = 128; s; s >>= 1) { if (tid < s) red[tid] += red[tid + s]; __syncthreads(); }
    float inv = 1.f / red[0]; __syncthreads();

    for (int n = tid; n < NTOK; n += 256) {
        size_t i = (size_t)n * DMODEL;
        base[i] = expf(base[i] - mx) * inv;
    }
}

// ctx[b,h,dk,e] = sum_n k[b,n,h,dk] * v[b,n,h,e]   (32x32 per (b,h))
__global__ __launch_bounds__(256)
void ctx_kernel(const float* __restrict__ k, const float* __restrict__ v,
                float* __restrict__ ctx)
{
    int bh = blockIdx.x;                  // 0..15
    int b = bh >> 3, h = bh & 7;
    int tid = threadIdx.x;
    int dk = tid >> 3;
    int e0 = (tid & 7) * 4;
    const float* kp = k + (size_t)b * NTOK * DMODEL + h * DKH + dk;
    const float* vp = v + (size_t)b * NTOK * DMODEL + h * DKH + e0;
    float a0 = 0.f, a1 = 0.f, a2 = 0.f, a3 = 0.f;
    for (int n = 0; n < NTOK; ++n) {
        float kv = kp[(size_t)n * DMODEL];
        float4 vv = *reinterpret_cast<const float4*>(vp + (size_t)n * DMODEL);
        a0 += kv * vv.x; a1 += kv * vv.y; a2 += kv * vv.z; a3 += kv * vv.w;
    }
    float* o = ctx + ((size_t)bh * 32 + dk) * 32 + e0;
    o[0] = a0; o[1] = a1; o[2] = a2; o[3] = a3;
}

// o[bt, h*32+e] = sum_dk q[bt, h*32+dk] * ctx[b,h,dk,e]  -> bf16 for Wo GEMM
__global__ void qctx_kernel(const float* __restrict__ q, const float* __restrict__ ctx,
                            bf16* __restrict__ obf)
{
    int i  = blockIdx.x * 256 + threadIdx.x;  // TBTOT*256
    int bt = i >> 8, c = i & 255;
    int h = c >> 5, e = c & 31;
    int b = bt / NTOK;
    const float* qp = q + (size_t)bt * DMODEL + h * DKH;
    const float* cp = ctx + ((size_t)(b * 8 + h) * 32) * 32 + e;
    float s = 0.f;
#pragma unroll
    for (int d = 0; d < 32; ++d) s += qp[d] * cp[d * 32];
    obf[i] = (bf16)s;
}

// t_out = LayerNorm(t + delta) * g + b ; optional bf16 mirror. One token/block.
__global__ __launch_bounds__(256)
void add_ln_kernel(const float* __restrict__ t, const float* __restrict__ delta,
                   const float* __restrict__ g, const float* __restrict__ bta,
                   float* __restrict__ tout, bf16* __restrict__ tbf)
{
    __shared__ float red[256];
    int bt = blockIdx.x, c = threadIdx.x;
    size_t idx = (size_t)bt * DMODEL + c;
    float r = t[idx] + delta[idx];
    red[c] = r; __syncthreads();
    for (int s = 128; s; s >>= 1) { if (c < s) red[c] += red[c + s]; __syncthreads(); }
    float mu = red[0] * (1.f / 256.f); __syncthreads();
    float dm = r - mu;
    red[c] = dm * dm; __syncthreads();
    for (int s = 128; s; s >>= 1) { if (c < s) red[c] += red[c + s]; __syncthreads(); }
    float var = red[0] * (1.f / 256.f);
    float y = dm * rsqrtf(var + 1e-6f) * g[c] + bta[c];
    tout[idx] = y;
    if (tbf) tbf[idx] = (bf16)y;
}

// x + depthwise3x3x3(x) + bias ; token-major in, optional NCDHW final output
__global__ void posconv_kernel(const float* __restrict__ t, const float* __restrict__ w,
                               const float* __restrict__ bias,
                               float* __restrict__ tout, bf16* __restrict__ tbf,
                               float* __restrict__ out_nc)
{
    int i  = blockIdx.x * 256 + threadIdx.x;  // NBATCH*NTOK*256
    int c  = i & 255;
    int nn = i >> 8;                          // b*NTOK + n
    int b  = nn / NTOK, n = nn - b * NTOK;
    int h = n / 576, rem = n - h * 576, wy = rem / 24, d = rem - wy * 24;
    const float* tb = t + (size_t)b * NTOK * DMODEL + c;
    const float* wc = w + c * 27;
    float acc = bias[c];
#pragma unroll
    for (int kd = 0; kd < 3; ++kd) {
        int hh = h + kd - 1;
        if ((unsigned)hh >= 24u) continue;
#pragma unroll
        for (int kh = 0; kh < 3; ++kh) {
            int ww = wy + kh - 1;
            if ((unsigned)ww >= 24u) continue;
#pragma unroll
            for (int kw = 0; kw < 3; ++kw) {
                int dd = d + kw - 1;
                if ((unsigned)dd >= 24u) continue;
                int np = (hh * 24 + ww) * 24 + dd;
                acc += wc[(kd * 3 + kh) * 3 + kw] * tb[(size_t)np * DMODEL];
            }
        }
    }
    float val = t[(size_t)nn * DMODEL + c] + acc;
    if (tout)  tout[(size_t)nn * DMODEL + c] = val;
    if (tbf)   tbf[(size_t)nn * DMODEL + c] = (bf16)val;
    if (out_nc) out_nc[((size_t)b * DMODEL + c) * NTOK + n] = val;
}

// ---------------------------------------------------------------------------
extern "C" void kernel_launch(void* const* d_in, const int* in_sizes, int n_in,
                              void* d_out, int out_size, void* d_ws, size_t ws_size,
                              hipStream_t stream)
{
    (void)in_sizes; (void)n_in; (void)out_size; (void)ws_size;
    const float* x    = (const float*)d_in[0];
    const float* Wx_w = (const float*)d_in[1];
    const float* Wx_b = (const float*)d_in[2];
    const float* Wq   = (const float*)d_in[3];
    const float* bq   = (const float*)d_in[4];
    const float* Wk   = (const float*)d_in[5];
    const float* bk   = (const float*)d_in[6];
    const float* Wv   = (const float*)d_in[7];
    const float* bv   = (const float*)d_in[8];
    const float* Wo   = (const float*)d_in[9];
    const float* bo   = (const float*)d_in[10];
    const float* ln1g = (const float*)d_in[11];
    const float* ln1b = (const float*)d_in[12];
    const float* ln2g = (const float*)d_in[13];
    const float* ln2b = (const float*)d_in[14];
    const float* W1   = (const float*)d_in[15];
    const float* b1   = (const float*)d_in[16];
    const float* W2   = (const float*)d_in[17];
    const float* b2   = (const float*)d_in[18];
    const float* pw   = (const float*)d_in[19];
    const float* pb   = (const float*)d_in[20];
    float* out = (float*)d_out;

    // workspace carve-up
    char* base = (char*)d_ws;
    size_t off = 0;
    auto alloc = [&](size_t bytes) -> void* {
        void* p = base + off;
        off = (off + bytes + 255) & ~(size_t)255;
        return p;
    };
    float* tA   = (float*)alloc((size_t)TBTOT * DMODEL * 4);
    float* tB   = (float*)alloc((size_t)TBTOT * DMODEL * 4);
    bf16*  tbf  = (bf16*) alloc((size_t)TBTOT * DMODEL * 2);
    float* qb   = (float*)alloc((size_t)TBTOT * DMODEL * 4);   // reused as o_proj
    float* kb   = (float*)alloc((size_t)TBTOT * DMODEL * 4);   // reused as ffn_out
    float* vb   = (float*)alloc((size_t)TBTOT * DMODEL * 4);
    bf16*  obf  = (bf16*) alloc((size_t)TBTOT * DMODEL * 2);
    bf16*  hbf  = (bf16*) alloc((size_t)TBTOT * DFFN   * 2);
    bf16*  xtb  = (bf16*) alloc((size_t)TBTOT * 128    * 2);
    float* ctxb = (float*)alloc((size_t)NBATCH * NHEAD * 32 * 32 * 4);
    bf16*  wxb  = (bf16*) alloc((size_t)256 * 128 * 2);        // [N=256][K=128]
    bf16*  wqb  = (bf16*) alloc((size_t)4 * 65536 * 2);        // [N][K] per layer
    bf16*  wkb  = (bf16*) alloc((size_t)4 * 65536 * 2);
    bf16*  wvb  = (bf16*) alloc((size_t)4 * 65536 * 2);
    bf16*  wob  = (bf16*) alloc((size_t)4 * 65536 * 2);
    bf16*  w1b  = (bf16*) alloc((size_t)4 * 131072 * 2);       // [512][256]
    bf16*  w2b  = (bf16*) alloc((size_t)4 * 131072 * 2);       // [256][512]

    // weight casts; all GEMM B operands stored as Wt = [N][K] bf16
    cast_f32_bf16<<<128, 256, 0, stream>>>(Wx_w, wxb, 256 * 128);  // already [out][in]
    for (int i = 0; i < 4; ++i) {
        cast_transpose<<<256, 256, 0, stream>>>(Wq + i * 65536,  wqb + i * 65536,  256, 256);
        cast_transpose<<<256, 256, 0, stream>>>(Wk + i * 65536,  wkb + i * 65536,  256, 256);
        cast_transpose<<<256, 256, 0, stream>>>(Wv + i * 65536,  wvb + i * 65536,  256, 256);
        cast_transpose<<<256, 256, 0, stream>>>(Wo + i * 65536,  wob + i * 65536,  256, 256);
        cast_transpose<<<512, 256, 0, stream>>>(W1 + i * 131072, w1b + i * 131072, 256, 512);
        cast_transpose<<<512, 256, 0, stream>>>(W2 + i * 131072, w2b + i * 131072, 512, 256);
    }

    // input projection: t = relu(x^T * Wx^T + b)
    transpose_cast_x<<<(TBTOT * 128) / 256, 256, 0, stream>>>(x, xtb);
    gemm_bf16_wmma<1><<<dim3(TBTOT / 64, 256 / 64), 128, 0, stream>>>(
        xtb, wxb, Wx_b, tA, tbf, TBTOT, 256, 128);

    float* tc = tA;
    float* tn = tB;
    for (int i = 0; i < 4; ++i) {
        const int wOff = i * 65536, bOff = i * 256;
        gemm_bf16_wmma<0><<<dim3(TBTOT / 64, 4), 128, 0, stream>>>(
            tbf, wqb + wOff, bq + bOff, qb, (bf16*)nullptr, TBTOT, 256, 256);
        gemm_bf16_wmma<0><<<dim3(TBTOT / 64, 4), 128, 0, stream>>>(
            tbf, wkb + wOff, bk + bOff, kb, (bf16*)nullptr, TBTOT, 256, 256);
        gemm_bf16_wmma<0><<<dim3(TBTOT / 64, 4), 128, 0, stream>>>(
            tbf, wvb + wOff, bv + bOff, vb, (bf16*)nullptr, TBTOT, 256, 256);

        softmax_q_kernel<<<TBTOT, 256, 0, stream>>>(qb);
        softmax_k_kernel<<<NBATCH * 256, 256, 0, stream>>>(kb);
        ctx_kernel<<<NBATCH * NHEAD, 256, 0, stream>>>(kb, vb, ctxb);
        qctx_kernel<<<(TBTOT * 256) / 256, 256, 0, stream>>>(qb, ctxb, obf);

        gemm_bf16_wmma<0><<<dim3(TBTOT / 64, 4), 128, 0, stream>>>(
            obf, wob + wOff, bo + bOff, qb /* o_proj */, (bf16*)nullptr, TBTOT, 256, 256);
        add_ln_kernel<<<TBTOT, 256, 0, stream>>>(tc, qb, ln1g + bOff, ln1b + bOff, tc, tbf);

        gemm_bf16_wmma<2><<<dim3(TBTOT / 64, 8), 128, 0, stream>>>(
            tbf, w1b + i * 131072, b1 + i * 512, (float*)nullptr, hbf, TBTOT, 512, 256);
        gemm_bf16_wmma<0><<<dim3(TBTOT / 64, 4), 128, 0, stream>>>(
            hbf, w2b + i * 131072, b2 + bOff, kb /* ffn_out */, (bf16*)nullptr, TBTOT, 256, 512);
        add_ln_kernel<<<TBTOT, 256, 0, stream>>>(tc, kb, ln2g + bOff, ln2b + bOff, tc, (bf16*)nullptr);

        const bool last = (i == 3);
        posconv_kernel<<<(TBTOT * 256) / 256, 256, 0, stream>>>(
            tc, pw + (size_t)i * 256 * 27, pb + bOff,
            last ? (float*)nullptr : tn,
            last ? (bf16*)nullptr : tbf,
            last ? out : (float*)nullptr);
        float* tmp = tc; tc = tn; tn = tmp;
    }
}
